// view_transformer_multi_scale_34961033789533
// MI455X (gfx1250) — compile-verified, matrix operand order checked by reading, not run on hardware
//
#include <hip/hip_runtime.h>
#include <math.h>

typedef __bf16 bf16_t;
typedef __attribute__((ext_vector_type(16))) __bf16 v16bf;
typedef __attribute__((ext_vector_type(8)))  float  v8f;

// Problem constants
constexpr int BATCH = 256;
constexpr int NV    = 40;
constexpr int DM    = 512;
constexpr int DKH   = 256;   // per-head dim (DK == DV)
constexpr int NH    = 4;
constexpr int DFF   = 2048;
constexpr int MROWS = BATCH * NV;       // 10240
constexpr int HDM   = NH * DKH;         // 1024
constexpr float EPSV = 1e-5f;
constexpr float NEG_BIG = -1e30f;

// ---------------------------------------------------------------------------
// KNN mask -> additive attention bias (0 for kept keys, -1e30 for masked)
// ---------------------------------------------------------------------------
__global__ __launch_bounds__(64) void knn_kernel(const float* __restrict__ verts,
                                                 float* __restrict__ bias1,
                                                 float* __restrict__ bias2) {
    int b = blockIdx.x;
    __shared__ float vx[NV], vy[NV], vz[NV];
    int t = threadIdx.x;
    if (t < NV) {
        vx[t] = verts[(b * NV + t) * 3 + 0];
        vy[t] = verts[(b * NV + t) * 3 + 1];
        vz[t] = verts[(b * NV + t) * 3 + 2];
    }
    __syncthreads();
    if (t < NV) {
        float d[NV];
        bool  sel[NV];
        for (int m = 0; m < NV; ++m) {
            float dx = vx[t] - vx[m], dy = vy[t] - vy[m], dz = vz[t] - vz[m];
            d[m] = dx * dx + dy * dy + dz * dz;
            sel[m] = false;
        }
        float* r1 = bias1 + ((size_t)b * NV + t) * NV;
        float* r2 = bias2 + ((size_t)b * NV + t) * NV;
        for (int m = 0; m < NV; ++m) { r1[m] = NEG_BIG; r2[m] = NEG_BIG; }
        for (int r = 0; r < 8; ++r) {       // K2 = 8 selections; first 4 also feed K1
            int best = 0; float bv = 3.0e38f;
            for (int m = 0; m < NV; ++m)
                if (!sel[m] && d[m] < bv) { bv = d[m]; best = m; }
            sel[best] = true;
            if (r < 4) r1[best] = 0.0f;
            r2[best] = 0.0f;
        }
    }
}

// ---------------------------------------------------------------------------
// f32 -> bf16 conversion (elementwise) and transpose-convert for weights
// ---------------------------------------------------------------------------
__global__ void cvt_bf16_kernel(const float* __restrict__ x, bf16_t* __restrict__ y, int n) {
    int i = blockIdx.x * blockDim.x + threadIdx.x;
    if (i < n) y[i] = (bf16_t)x[i];
}

// W is [K,N] row-major f32; Wt is [N,K] row-major bf16
__global__ void tcvt_kernel(const float* __restrict__ W, bf16_t* __restrict__ Wt, int K, int N) {
    int i = blockIdx.x * blockDim.x + threadIdx.x;
    if (i < K * N) {
        int k = i / N, n = i % N;
        Wt[(size_t)n * K + k] = (bf16_t)W[i];
    }
}

// ---------------------------------------------------------------------------
// bf16 WMMA GEMM: C[M,N] = A[M,K] * Bt[N,K]^T + bias
// block = 256 threads (8 waves); block tile 128x64; wave tile 16x64 (4 WMMAs)
// epi==0: store f32 to Cf.  epi==1: relu, store bf16 to Cb.
// ---------------------------------------------------------------------------
union FragU { uint4 u[2]; v16bf v; };

__global__ __launch_bounds__(256) void gemm_bf16_kernel(
    const bf16_t* __restrict__ A, int lda,
    const bf16_t* __restrict__ Bt, int ldb,
    const float* __restrict__ bias,
    float* __restrict__ Cf, bf16_t* __restrict__ Cb,
    int N, int K, int epi)
{
    int wave = threadIdx.x >> 5;
    int lane = threadIdx.x & 31;
    int r0 = blockIdx.y * 128 + wave * 16;
    int c0 = blockIdx.x * 64;
    int mr = lane & 15;    // row (A) / col (B) within tile
    int hf = lane >> 4;    // half-wave selects K sub-pattern

    const bf16_t* pa  = A  + (size_t)(r0 + mr) * lda + hf * 8;
    const bf16_t* pb0 = Bt + (size_t)(c0 +  0 + mr) * ldb + hf * 8;
    const bf16_t* pb1 = Bt + (size_t)(c0 + 16 + mr) * ldb + hf * 8;
    const bf16_t* pb2 = Bt + (size_t)(c0 + 32 + mr) * ldb + hf * 8;
    const bf16_t* pb3 = Bt + (size_t)(c0 + 48 + mr) * ldb + hf * 8;

    v8f acc[4];
#pragma unroll
    for (int j = 0; j < 4; ++j) acc[j] = (v8f){0.f,0.f,0.f,0.f,0.f,0.f,0.f,0.f};

    for (int kk = 0; kk < K; kk += 32) {
        FragU fa, fb0, fb1, fb2, fb3;
        fa.u[0]  = *(const uint4*)(pa  + kk);
        fa.u[1]  = *(const uint4*)(pa  + kk + 16);
        fb0.u[0] = *(const uint4*)(pb0 + kk);
        fb0.u[1] = *(const uint4*)(pb0 + kk + 16);
        fb1.u[0] = *(const uint4*)(pb1 + kk);
        fb1.u[1] = *(const uint4*)(pb1 + kk + 16);
        fb2.u[0] = *(const uint4*)(pb2 + kk);
        fb2.u[1] = *(const uint4*)(pb2 + kk + 16);
        fb3.u[0] = *(const uint4*)(pb3 + kk);
        fb3.u[1] = *(const uint4*)(pb3 + kk + 16);

        acc[0] = __builtin_amdgcn_wmma_f32_16x16x32_bf16(false, fa.v, false, fb0.v, (short)0, acc[0], false, false);
        acc[1] = __builtin_amdgcn_wmma_f32_16x16x32_bf16(false, fa.v, false, fb1.v, (short)0, acc[1], false, false);
        acc[2] = __builtin_amdgcn_wmma_f32_16x16x32_bf16(false, fa.v, false, fb2.v, (short)0, acc[2], false, false);
        acc[3] = __builtin_amdgcn_wmma_f32_16x16x32_bf16(false, fa.v, false, fb3.v, (short)0, acc[3], false, false);
    }

#pragma unroll
    for (int j = 0; j < 4; ++j) {
        int col = c0 + j * 16 + mr;
        float bs = bias ? bias[col] : 0.0f;
#pragma unroll
        for (int v = 0; v < 8; ++v) {
            int row = r0 + hf * 8 + v;
            float val = acc[j][v] + bs;
            size_t o = (size_t)row * N + col;
            if (epi == 1) {
                val = fmaxf(val, 0.0f);
                Cb[o] = (bf16_t)val;
            } else {
                Cf[o] = val;
            }
        }
    }
}

// ---------------------------------------------------------------------------
// Masked attention per (b,h): scores in f32, softmax, output stored as bf16
// ---------------------------------------------------------------------------
__global__ __launch_bounds__(256) void attn_kernel(
    const float* __restrict__ Q, const float* __restrict__ K,
    const float* __restrict__ V, const float* __restrict__ bias,
    bf16_t* __restrict__ Obf)
{
    constexpr int KST = 264;  // padded LDS row stride (floats)
    int bh = blockIdx.x;
    int b = bh / NH, h = bh % NH;
    int t = threadIdx.x;

    __shared__ float Ksh[NV * KST];
    __shared__ float qsh[DKH];
    __shared__ float psh[64];
    __shared__ float esh[64];

    for (int idx = t; idx < NV * DKH; idx += 256) {
        int j = idx >> 8, d = idx & 255;
        Ksh[j * KST + d] = K[((size_t)(b * NV + j)) * HDM + h * DKH + d];
    }
    __syncthreads();

    for (int i = 0; i < NV; ++i) {
        qsh[t] = Q[((size_t)(b * NV + i)) * HDM + h * DKH + t];
        __syncthreads();
        if (t < NV) {
            float s = 0.0f;
            const float* kr = &Ksh[t * KST];
            for (int d = 0; d < DKH; ++d) s += qsh[d] * kr[d];
            psh[t] = s * 0.0625f + bias[((size_t)(b * NV + i)) * NV + t];
        }
        __syncthreads();
        float mx = -3.0e38f;
        for (int j = 0; j < NV; ++j) mx = fmaxf(mx, psh[j]);
        if (t < NV) esh[t] = expf(psh[t] - mx);
        __syncthreads();
        float den = 0.0f;
        for (int j = 0; j < NV; ++j) den += esh[j];
        float o = 0.0f;
        for (int j = 0; j < NV; ++j)
            o += esh[j] * V[((size_t)(b * NV + j)) * HDM + h * DKH + t];
        Obf[((size_t)(b * NV + i)) * HDM + h * DKH + t] = (bf16_t)(o / den);
        __syncthreads();
    }
}

// ---------------------------------------------------------------------------
// LayerNorm over last dim (512) of (X + R); optional f32 and bf16 outputs
// ---------------------------------------------------------------------------
__global__ __launch_bounds__(256) void ln_kernel(
    const float* __restrict__ X, const float* __restrict__ R,
    const float* __restrict__ g, const float* __restrict__ beta,
    float* __restrict__ outF, bf16_t* __restrict__ outB,
    int obStride, int obOff)
{
    int row = blockIdx.x;
    int t = threadIdx.x;
    float v0 = X[(size_t)row * DM + t]       + R[(size_t)row * DM + t];
    float v1 = X[(size_t)row * DM + t + 256] + R[(size_t)row * DM + t + 256];

    __shared__ float red[256];
    __shared__ float sMean, sVar;
    red[t] = v0 + v1;
    __syncthreads();
    for (int k = 128; k > 0; k >>= 1) { if (t < k) red[t] += red[t + k]; __syncthreads(); }
    if (t == 0) sMean = red[0] / (float)DM;
    __syncthreads();
    red[t] = v0 * v0 + v1 * v1;
    __syncthreads();
    for (int k = 128; k > 0; k >>= 1) { if (t < k) red[t] += red[t + k]; __syncthreads(); }
    if (t == 0) sVar = red[0] / (float)DM - sMean * sMean;
    __syncthreads();

    float inv = rsqrtf(sVar + EPSV);
    float y0 = (v0 - sMean) * inv * g[t]       + beta[t];
    float y1 = (v1 - sMean) * inv * g[t + 256] + beta[t + 256];
    if (outF) {
        outF[(size_t)row * DM + t] = y0;
        outF[(size_t)row * DM + t + 256] = y1;
    }
    if (outB) {
        outB[(size_t)row * obStride + obOff + t] = (bf16_t)y0;
        outB[(size_t)row * obStride + obOff + t + 256] = (bf16_t)y1;
    }
}

// ---------------------------------------------------------------------------
// BatchNorm stats (per channel over all rows; biased variance) + apply+LeakyReLU
// ---------------------------------------------------------------------------
__global__ __launch_bounds__(256) void bn_stats_kernel(const float* __restrict__ Z,
                                                       float* __restrict__ stats) {
    int c = blockIdx.x;
    int t = threadIdx.x;
    float s = 0.0f, s2 = 0.0f;
    for (int r = t; r < MROWS; r += 256) {
        float v = Z[(size_t)r * DM + c];
        s += v; s2 += v * v;
    }
    __shared__ float rs[256], rq[256];
    rs[t] = s; rq[t] = s2;
    __syncthreads();
    for (int k = 128; k > 0; k >>= 1) {
        if (t < k) { rs[t] += rs[t + k]; rq[t] += rq[t + k]; }
        __syncthreads();
    }
    if (t == 0) {
        float m = rs[0] / (float)MROWS;
        stats[c]      = m;
        stats[DM + c] = rq[0] / (float)MROWS - m * m;
    }
}

__global__ void bn_apply_kernel(const float* __restrict__ Z, const float* __restrict__ stats,
                                const float* __restrict__ g, const float* __restrict__ bb,
                                bf16_t* __restrict__ out, int n) {
    int i = blockIdx.x * blockDim.x + threadIdx.x;
    if (i < n) {
        int c = i & (DM - 1);
        float y = (Z[i] - stats[c]) * rsqrtf(stats[DM + c] + EPSV) * g[c] + bb[c];
        y = (y > 0.0f) ? y : 0.2f * y;
        out[i] = (bf16_t)y;
    }
}

// ---------------------------------------------------------------------------
// Host-side launch
// ---------------------------------------------------------------------------
static inline void launch_gemm(hipStream_t s, const bf16_t* A, int lda,
                               const bf16_t* Bt, int ldb, const float* bias,
                               float* Cf, bf16_t* Cb, int M, int N, int K, int epi) {
    dim3 grid(N / 64, M / 128);
    gemm_bf16_kernel<<<grid, 256, 0, s>>>(A, lda, Bt, ldb, bias, Cf, Cb, N, K, epi);
}

extern "C" void kernel_launch(void* const* d_in, const int* in_sizes, int n_in,
                              void* d_out, int out_size, void* d_ws, size_t ws_size,
                              hipStream_t stream) {
    (void)in_sizes; (void)n_in; (void)out_size; (void)ws_size;
    const float* verts = (const float*)d_in[0];
    const float* F     = (const float*)d_in[1];

    char* ws = (char*)d_ws;
    size_t off = 0;
    auto alloc = [&](size_t bytes) -> char* {
        char* p = ws + off;
        off += (bytes + 255) & ~(size_t)255;
        return p;
    };

    bf16_t* Fbf   = (bf16_t*)alloc((size_t)MROWS * DM * 2);
    float*  bias1 = (float*) alloc((size_t)BATCH * NV * NV * 4);
    float*  bias2 = (float*) alloc((size_t)BATCH * NV * NV * 4);
    float*  Qf    = (float*) alloc((size_t)MROWS * HDM * 4);
    float*  Kf    = (float*) alloc((size_t)MROWS * HDM * 4);
    float*  Vf    = (float*) alloc((size_t)MROWS * HDM * 4);
    bf16_t* Obf   = (bf16_t*)alloc((size_t)MROWS * HDM * 2);
    float*  Pf    = (float*) alloc((size_t)MROWS * DM * 4);
    float*  X1f   = (float*) alloc((size_t)MROWS * DM * 4);
    bf16_t* X1bf  = (bf16_t*)alloc((size_t)MROWS * DM * 2);
    bf16_t* Hbf   = (bf16_t*)alloc((size_t)MROWS * DFF * 2);
    float*  Gf    = (float*) alloc((size_t)MROWS * DM * 4);
    bf16_t* Ybf   = (bf16_t*)alloc((size_t)MROWS * HDM * 2);
    float*  Zf    = (float*) alloc((size_t)MROWS * DM * 4);
    bf16_t* Znbf  = (bf16_t*)alloc((size_t)MROWS * DM * 2);
    float*  stats = (float*) alloc((size_t)2 * DM * 4);

    bf16_t *wt_q[2], *wt_k[2], *wt_v[2], *wt_o[2], *wt_1[2], *wt_2[2];
    for (int e = 0; e < 2; ++e) {
        wt_q[e] = (bf16_t*)alloc((size_t)DM * HDM * 2);
        wt_k[e] = (bf16_t*)alloc((size_t)DM * HDM * 2);
        wt_v[e] = (bf16_t*)alloc((size_t)DM * HDM * 2);
        wt_o[e] = (bf16_t*)alloc((size_t)HDM * DM * 2);
        wt_1[e] = (bf16_t*)alloc((size_t)DM * DFF * 2);
        wt_2[e] = (bf16_t*)alloc((size_t)DFF * DM * 2);
    }
    bf16_t* wt_f1 = (bf16_t*)alloc((size_t)HDM * DM * 2);
    bf16_t* wt_f2 = (bf16_t*)alloc((size_t)DM * DM * 2);

    // --- Preprocessing -----------------------------------------------------
    {
        int n = MROWS * DM;
        cvt_bf16_kernel<<<(n + 255) / 256, 256, 0, stream>>>(F, Fbf, n);
    }
    knn_kernel<<<BATCH, 64, 0, stream>>>(verts, bias1, bias2);

    auto tcvt = [&](const void* W, bf16_t* Wt, int K, int N) {
        int n = K * N;
        tcvt_kernel<<<(n + 255) / 256, 256, 0, stream>>>((const float*)W, Wt, K, N);
    };
    for (int e = 0; e < 2; ++e) {
        int i0 = 2 + 16 * e;
        tcvt(d_in[i0 + 0],  wt_q[e], DM,  HDM);
        tcvt(d_in[i0 + 2],  wt_k[e], DM,  HDM);
        tcvt(d_in[i0 + 4],  wt_v[e], DM,  HDM);
        tcvt(d_in[i0 + 6],  wt_o[e], HDM, DM);
        tcvt(d_in[i0 + 10], wt_1[e], DM,  DFF);
        tcvt(d_in[i0 + 12], wt_2[e], DFF, DM);
    }
    tcvt(d_in[34], wt_f1, HDM, DM);
    tcvt(d_in[38], wt_f2, DM,  DM);

    // --- Two encoders ------------------------------------------------------
    for (int e = 0; e < 2; ++e) {
        int i0 = 2 + 16 * e;
        const float* bq   = (const float*)d_in[i0 + 1];
        const float* bk   = (const float*)d_in[i0 + 3];
        const float* bv   = (const float*)d_in[i0 + 5];
        const float* bo   = (const float*)d_in[i0 + 7];
        const float* ln1g = (const float*)d_in[i0 + 8];
        const float* ln1b = (const float*)d_in[i0 + 9];
        const float* b1   = (const float*)d_in[i0 + 11];
        const float* b2   = (const float*)d_in[i0 + 13];
        const float* ln2g = (const float*)d_in[i0 + 14];
        const float* ln2b = (const float*)d_in[i0 + 15];
        const float* abias = (e == 0) ? bias1 : bias2;

        launch_gemm(stream, Fbf, DM, wt_q[e], DM, bq, Qf, nullptr, MROWS, HDM, DM, 0);
        launch_gemm(stream, Fbf, DM, wt_k[e], DM, bk, Kf, nullptr, MROWS, HDM, DM, 0);
        launch_gemm(stream, Fbf, DM, wt_v[e], DM, bv, Vf, nullptr, MROWS, HDM, DM, 0);

        attn_kernel<<<BATCH * NH, 256, 0, stream>>>(Qf, Kf, Vf, abias, Obf);

        launch_gemm(stream, Obf, HDM, wt_o[e], HDM, bo, Pf, nullptr, MROWS, DM, HDM, 0);
        ln_kernel<<<MROWS, 256, 0, stream>>>(F, Pf, ln1g, ln1b, X1f, X1bf, DM, 0);

        launch_gemm(stream, X1bf, DM, wt_1[e], DM, b1, nullptr, Hbf, MROWS, DFF, DM, 1);
        launch_gemm(stream, Hbf, DFF, wt_2[e], DFF, b2, Gf, nullptr, MROWS, DM, DFF, 0);

        ln_kernel<<<MROWS, 256, 0, stream>>>(X1f, Gf, ln2g, ln2b, nullptr, Ybf, HDM, e * DM);
    }

    // --- Fusion ------------------------------------------------------------
    const float* fb1 = (const float*)d_in[35];
    const float* bng = (const float*)d_in[36];
    const float* bnb = (const float*)d_in[37];
    const float* fb2 = (const float*)d_in[39];

    launch_gemm(stream, Ybf, HDM, wt_f1, HDM, fb1, Zf, nullptr, MROWS, DM, HDM, 0);
    bn_stats_kernel<<<DM, 256, 0, stream>>>(Zf, stats);
    {
        int n = MROWS * DM;
        bn_apply_kernel<<<(n + 255) / 256, 256, 0, stream>>>(Zf, stats, bng, bnb, Znbf, n);
    }
    launch_gemm(stream, Znbf, DM, wt_f2, DM, fb2, (float*)d_out, nullptr, MROWS, DM, DM, 0);
}
